// Seq2SeqModel_42941083025755
// MI455X (gfx1250) — compile-verified, matrix operand order checked by reading, not run on hardware
//
#include <hip/hip_runtime.h>
#include <math.h>

typedef unsigned short u16;
typedef __attribute__((ext_vector_type(16))) __bf16 v16bf;
typedef __attribute__((ext_vector_type(8)))  __bf16 v8bf;
typedef __attribute__((ext_vector_type(8)))  float  v8f;

#define Bsz 1024
#define S_ENC 49
#define T_DEC 25
#define Idim 76
#define Hdim 2048
#define G3 6144          // 3*H
#define KPAD_I 96        // 76 padded to multiple of 32
#define NPAD_FC 80       // 76 padded to multiple of 16

union ABfrag { v16bf v; v8bf h[2]; };

__device__ __forceinline__ u16 f2bf(float f) {
    unsigned u = __float_as_uint(f);
    unsigned r = u + 0x7FFFu + ((u >> 16) & 1u);   // round-to-nearest-even
    return (u16)(r >> 16);
}

// ---------------------------------------------------------------------------
// Convert fp32 [Nsrc][Ksrc] -> bf16 [Npad][Kpad], zero padded.
__global__ void conv_pad_bf16(const float* __restrict__ src, u16* __restrict__ dst,
                              int Nsrc, int Ksrc, int Npad, int Kpad) {
    int idx = blockIdx.x * blockDim.x + threadIdx.x;
    if (idx >= Npad * Kpad) return;
    int n = idx / Kpad, k = idx - n * Kpad;
    float v = (n < Nsrc && k < Ksrc) ? src[(size_t)n * Ksrc + k] : 0.0f;
    dst[idx] = f2bf(v);
}

// encoder_inputs [B][S][I] fp32 -> [S][B][KPAD_I] bf16 (time-major slices)
__global__ void conv_enc(const float* __restrict__ enc, u16* __restrict__ dst) {
    int idx = blockIdx.x * blockDim.x + threadIdx.x;
    if (idx >= S_ENC * Bsz * KPAD_I) return;
    int k = idx % KPAD_I;
    int r = idx / KPAD_I;
    int b = r % Bsz, t = r / Bsz;
    float v = (k < Idim) ? enc[((size_t)b * S_ENC + t) * Idim + k] : 0.0f;
    dst[idx] = f2bf(v);
}

// decoder_inputs[:,0,:] -> x_bf16 [B][KPAD_I] + x_f32 [B][I]
__global__ void conv_dec0(const float* __restrict__ dec, u16* __restrict__ xb,
                          float* __restrict__ xf) {
    int idx = blockIdx.x * blockDim.x + threadIdx.x;
    if (idx >= Bsz * KPAD_I) return;
    int k = idx % KPAD_I, b = idx / KPAD_I;
    float v = (k < Idim) ? dec[(size_t)b * T_DEC * Idim + k] : 0.0f;
    xb[idx] = f2bf(v);
    if (k < Idim) xf[(size_t)b * Idim + k] = v;
}

__global__ void zero_h(float* __restrict__ h, u16* __restrict__ hb) {
    int idx = blockIdx.x * blockDim.x + threadIdx.x;
    if (idx >= Bsz * Hdim) return;
    h[idx] = 0.0f;
    hb[idx] = 0;
}

// ---------------------------------------------------------------------------
// Fragment helpers (everything fully unrolled; no divergent control flow).
template<int T>
__device__ __forceinline__ void load_frags(ABfrag (&f)[T], const u16* const (&p)[T], int k0) {
#pragma unroll
    for (int i = 0; i < T; ++i) {
        f[i].h[0] = *reinterpret_cast<const v8bf*>(p[i] + k0);
        f[i].h[1] = *reinterpret_cast<const v8bf*>(p[i] + k0 + 16);
    }
}

template<int MT, int NT>
__device__ __forceinline__ void mma_all(v8f (&acc)[MT][NT],
                                        const ABfrag (&a)[MT], const ABfrag (&b)[NT]) {
#pragma unroll
    for (int i = 0; i < MT; ++i)
#pragma unroll
        for (int j = 0; j < NT; ++j)
            acc[i][j] = __builtin_amdgcn_wmma_f32_16x16x32_bf16(
                false, a[i].v, false, b[j].v, (short)0, acc[i][j], false, false);
}

// bf16 WMMA GEMM: C[M][N] (fp32, ldc) = A[M][K] (bf16, lda) x B^T, with B stored
// row-major [N][K] (ldb). One wave computes a (MT*16)x(NT*16) tile.
// Requires: M % (16*MT) == 0, N % (16*NT) == 0, K % 32 == 0.
template<int MT, int NT>
__global__ __launch_bounds__(256) void gemm_wmma(
    const u16* __restrict__ A, int lda,
    const u16* __restrict__ B, int ldb,
    float* __restrict__ C, int ldc,
    int M, int N, int K) {

    const int numMt = M / (16 * MT);
    const int numNs = N / (16 * NT);
    const int wave  = blockIdx.x * (blockDim.x >> 5) + (threadIdx.x >> 5);
    if (wave >= numMt * numNs) return;
    const int lane  = threadIdx.x & 31;
    const int mt = wave % numMt;
    const int ns = wave / numMt;
    const int m0 = mt * 16 * MT;
    const int n0 = ns * 16 * NT;

    const int lrow  = lane & 15;
    const int khalf = (lane >> 4) << 3;   // 0 (lanes 0-15) or 8 (lanes 16-31)

    const u16* aPtr[MT];
    const u16* bPtr[NT];
#pragma unroll
    for (int i = 0; i < MT; ++i)
        aPtr[i] = A + (size_t)(m0 + i * 16 + lrow) * lda + khalf;
#pragma unroll
    for (int j = 0; j < NT; ++j)
        bPtr[j] = B + (size_t)(n0 + j * 16 + lrow) * ldb + khalf;

    v8f acc[MT][NT] = {};

    // Ping-pong double buffering: load chunk k+32 while doing WMMAs on chunk k.
    ABfrag a0[MT], b0[NT], a1[MT], b1[NT];
    load_frags<MT>(a0, aPtr, 0);
    load_frags<NT>(b0, bPtr, 0);

    int k0 = 0;
    while (k0 + 64 <= K) {
        load_frags<MT>(a1, aPtr, k0 + 32);
        load_frags<NT>(b1, bPtr, k0 + 32);
        mma_all<MT, NT>(acc, a0, b0);
        if (k0 + 64 < K) {
            load_frags<MT>(a0, aPtr, k0 + 64);
            load_frags<NT>(b0, bPtr, k0 + 64);
        }
        mma_all<MT, NT>(acc, a1, b1);
        k0 += 64;
    }
    if (k0 < K)                       // odd number of K-chunks: tail is in a0/b0
        mma_all<MT, NT>(acc, a0, b0);

    // C/D layout: VGPR elem e -> row m0 + mi*16 + 8*hi + e ; col = tile + (lane&15)
#pragma unroll
    for (int i = 0; i < MT; ++i)
#pragma unroll
        for (int j = 0; j < NT; ++j) {
            float* crow = C + (size_t)(m0 + i * 16 + khalf) * ldc + (n0 + j * 16 + lrow);
#pragma unroll
            for (int e = 0; e < 8; ++e)
                crow[(size_t)e * ldc] = acc[i][j][e];
        }
}

// ---------------------------------------------------------------------------
// GRU gate fusion: h_new = (1-z)*n + z*h ; writes h (fp32) + h (bf16)
__global__ void gru_gates(const float* __restrict__ gi, const float* __restrict__ gh,
                          const float* __restrict__ b_ih, const float* __restrict__ b_hh,
                          float* __restrict__ h, u16* __restrict__ hb) {
    int idx = blockIdx.x * blockDim.x + threadIdx.x;
    if (idx >= Bsz * Hdim) return;
    int b = idx >> 11;            // /2048
    int j = idx & (Hdim - 1);
    size_t g = (size_t)b * G3;
    float ir  = gi[g + j]             + b_ih[j];
    float iz  = gi[g + Hdim + j]      + b_ih[Hdim + j];
    float in_ = gi[g + 2 * Hdim + j]  + b_ih[2 * Hdim + j];
    float hr  = gh[g + j]             + b_hh[j];
    float hz  = gh[g + Hdim + j]      + b_hh[Hdim + j];
    float hn  = gh[g + 2 * Hdim + j]  + b_hh[2 * Hdim + j];
    float r = 1.0f / (1.0f + __expf(-(ir + hr)));
    float z = 1.0f / (1.0f + __expf(-(iz + hz)));
    float n = tanhf(in_ + r * hn);
    float hv = (1.0f - z) * n + z * h[idx];
    h[idx]  = hv;
    hb[idx] = f2bf(hv);
}

// out[b,t,:] = x + fcout + fc_b ; feed next step's input (fp32 + bf16)
__global__ void fc_add_store(const float* __restrict__ fcout, const float* __restrict__ fc_b,
                             float* __restrict__ xf, u16* __restrict__ xb,
                             float* __restrict__ out, int t) {
    int idx = blockIdx.x * blockDim.x + threadIdx.x;
    if (idx >= Bsz * KPAD_I) return;
    int k = idx % KPAD_I, b = idx / KPAD_I;
    if (k < Idim) {
        float v = xf[(size_t)b * Idim + k] + fcout[(size_t)b * NPAD_FC + k] + fc_b[k];
        out[((size_t)b * T_DEC + t) * Idim + k] = v;
        xf[(size_t)b * Idim + k] = v;
        xb[idx] = f2bf(v);
    } else {
        xb[idx] = 0;
    }
}

// ---------------------------------------------------------------------------
template<int MT, int NT>
static inline void launch_gemm(const u16* A, int lda, const u16* B, int ldb,
                               float* C, int ldc, int M, int N, int K,
                               hipStream_t s) {
    int waves  = (M / (16 * MT)) * (N / (16 * NT));
    int blocks = (waves + 7) >> 3;          // 8 waves (256 threads) per block
    gemm_wmma<MT, NT><<<blocks, 256, 0, s>>>(A, lda, B, ldb, C, ldc, M, N, K);
}

extern "C" void kernel_launch(void* const* d_in, const int* in_sizes, int n_in,
                              void* d_out, int out_size, void* d_ws, size_t ws_size,
                              hipStream_t stream) {
    const float* enc  = (const float*)d_in[0];
    const float* dec  = (const float*)d_in[1];
    const float* W_ih = (const float*)d_in[2];
    const float* W_hh = (const float*)d_in[3];
    const float* b_ih = (const float*)d_in[4];
    const float* b_hh = (const float*)d_in[5];
    const float* fc_w = (const float*)d_in[6];
    const float* fc_b = (const float*)d_in[7];
    float* out = (float*)d_out;

    char* ws = (char*)d_ws;
    size_t off = 0;
    auto alloc = [&](size_t bytes) -> void* {
        off = (off + 255) & ~(size_t)255;
        void* p = ws + off;
        off += bytes;
        return p;
    };
    u16*   Whh_b = (u16*)  alloc((size_t)G3 * Hdim * 2);            // 25.2 MB
    u16*   Wih_b = (u16*)  alloc((size_t)G3 * KPAD_I * 2);          //  1.2 MB
    u16*   fcw_b = (u16*)  alloc((size_t)NPAD_FC * Hdim * 2);       //  0.3 MB
    u16*   Xe    = (u16*)  alloc((size_t)S_ENC * Bsz * KPAD_I * 2); //  9.6 MB
    float* h     = (float*)alloc((size_t)Bsz * Hdim * 4);           //  8.4 MB
    u16*   hb    = (u16*)  alloc((size_t)Bsz * Hdim * 2);           //  4.2 MB
    float* gi    = (float*)alloc((size_t)Bsz * G3 * 4);             // 25.2 MB
    float* gh    = (float*)alloc((size_t)Bsz * G3 * 4);             // 25.2 MB
    u16*   xb    = (u16*)  alloc((size_t)Bsz * KPAD_I * 2);
    float* xf    = (float*)alloc((size_t)Bsz * Idim * 4);
    float* fco   = (float*)alloc((size_t)Bsz * NPAD_FC * 4);

    const int TB = 256;
    auto nblk = [](long n, int tb) { return (int)((n + tb - 1) / tb); };

    // One-time conversions (weights -> bf16 [n][k], inputs -> bf16 time slices)
    conv_pad_bf16<<<nblk((long)G3 * Hdim, TB), TB, 0, stream>>>(W_hh, Whh_b, G3, Hdim, G3, Hdim);
    conv_pad_bf16<<<nblk((long)G3 * KPAD_I, TB), TB, 0, stream>>>(W_ih, Wih_b, G3, Idim, G3, KPAD_I);
    conv_pad_bf16<<<nblk((long)NPAD_FC * Hdim, TB), TB, 0, stream>>>(fc_w, fcw_b, Idim, Hdim, NPAD_FC, Hdim);
    conv_enc<<<nblk((long)S_ENC * Bsz * KPAD_I, TB), TB, 0, stream>>>(enc, Xe);
    conv_dec0<<<nblk((long)Bsz * KPAD_I, TB), TB, 0, stream>>>(dec, xb, xf);
    zero_h<<<nblk((long)Bsz * Hdim, TB), TB, 0, stream>>>(h, hb);

    const int gateBlocks = nblk((long)Bsz * Hdim, TB);

    // Encoder: 49 serial GRU steps
    for (int t = 0; t < S_ENC; ++t) {
        const u16* xt = Xe + (size_t)t * Bsz * KPAD_I;
        launch_gemm<2, 4>(xt, KPAD_I, Wih_b, KPAD_I, gi, G3, Bsz, G3, KPAD_I, stream);
        launch_gemm<2, 4>(hb, Hdim,   Whh_b, Hdim,   gh, G3, Bsz, G3, Hdim,   stream);
        gru_gates<<<gateBlocks, TB, 0, stream>>>(gi, gh, b_ih, b_hh, h, hb);
    }

    // Decoder: 25 serial GRU + FC steps, output feeds next input
    for (int t = 0; t < T_DEC; ++t) {
        launch_gemm<2, 4>(xb, KPAD_I, Wih_b, KPAD_I, gi, G3, Bsz, G3, KPAD_I, stream);
        launch_gemm<2, 4>(hb, Hdim,   Whh_b, Hdim,   gh, G3, Bsz, G3, Hdim,   stream);
        gru_gates<<<gateBlocks, TB, 0, stream>>>(gi, gh, b_ih, b_hh, h, hb);
        launch_gemm<2, 1>(hb, Hdim, fcw_b, Hdim, fco, NPAD_FC, Bsz, NPAD_FC, Hdim, stream);
        fc_add_store<<<nblk((long)Bsz * KPAD_I, TB), TB, 0, stream>>>(fco, fc_b, xf, xb, out, t);
    }
}
